// YuaSparseMoE_61881888800983
// MI455X (gfx1250) — compile-verified, matrix operand order checked by reading, not run on hardware
//
#include <hip/hip_runtime.h>

// ---------------------------------------------------------------------------
// MoE (Mixtral-style, top-2 of 8 + shared expert) for MI455X / gfx1250.
// bf16 WMMA (v_wmma_f32_16x16x32_bf16), gather-based sparse routing,
// double-buffered LDS tiles fed by GLOBAL_LOAD_ASYNC_TO_LDS_B128 (ASYNCcnt).
// ---------------------------------------------------------------------------

#define HH 2048   // hidden
#define II 1024   // intermediate
#define EE 8      // routed experts
#define TT 8192   // tokens (4 * 2048)

typedef __bf16 v16bf __attribute__((ext_vector_type(16)));
typedef float  v8f   __attribute__((ext_vector_type(8)));

// LDS row stride (bf16 elems): 32 data + 8 pad -> 80B rows, conflict-free
// across the 16 rows a fragment load touches, 16B-aligned sub-chunks.
#define LDP 40

// ---- async global -> LDS 16B tile chunk (ASYNCcnt-tracked) ----------------
__device__ __forceinline__ void asyncLoad16(const __bf16* g, __bf16* l) {
  unsigned loff = (unsigned)(uintptr_t)l;  // low 32 bits = LDS byte offset
  asm volatile("global_load_async_to_lds_b128 %0, %1, off"
               :: "v"(loff), "v"(g) : "memory");
}
__device__ __forceinline__ void waitAsync0() {
  asm volatile("s_wait_asynccnt 0" ::: "memory");
}

// ---- fragment load: 16x32 bf16 A (or N-major B) tile row -> v16bf ---------
// ISA layout (CDNA5 ISA): lanes 0-15 = M rows, khalf=lane>>4 selects K-halves;
// per lane the 16 elements are two contiguous 8-elem K runs:
//   [8*khalf, 8*khalf+8) and [16+8*khalf, 24+8*khalf)
__device__ __forceinline__ v16bf ldsFrag(const __bf16* rowBase, int khalf) {
  struct alignas(16) P { uint4 lo, hi; } p;
  p.lo = *reinterpret_cast<const uint4*>(rowBase + (khalf << 3));
  p.hi = *reinterpret_cast<const uint4*>(rowBase + 16 + (khalf << 3));
  return __builtin_bit_cast(v16bf, p);
}

// ---------------------------------------------------------------------------
// small utility kernels
// ---------------------------------------------------------------------------
__global__ void zero_f32(float* __restrict__ p, int n) {
  int i = blockIdx.x * blockDim.x + threadIdx.x;
  if (i < n) p[i] = 0.0f;
}

__global__ void init_counts(int* __restrict__ counts) {
  if (threadIdx.x == 0) {
    #pragma unroll
    for (int e = 0; e < EE; ++e) counts[e] = 0;
    counts[EE] = TT;  // shared "expert" processes every token
  }
}

__global__ void convert_x(const float* __restrict__ x, __bf16* __restrict__ xb) {
  int i = blockIdx.x * blockDim.x + threadIdx.x;
  xb[i] = (__bf16)x[i];  // grid sized exactly TT*HH
}

__global__ void shared_list(int* __restrict__ idxL, float* __restrict__ wgtL) {
  int t = blockIdx.x * blockDim.x + threadIdx.x;
  if (t < TT) {
    idxL[EE * TT + t] = t;
    wgtL[EE * TT + t] = 1.0f;
  }
}

// fp32 [R,C] -> bf16 [C,R] tiled transpose-convert (weights become N-major so
// B tiles in the GEMMs load with the same contiguous pattern as A tiles).
__global__ __launch_bounds__(256) void tconv(const float* __restrict__ src,
                                             __bf16* __restrict__ dst,
                                             int R, int C) {
  __shared__ float t[32][33];
  int bx = blockIdx.x * 32, by = blockIdx.y * 32;
  int tx = threadIdx.x, ty = threadIdx.y;  // block (32,8)
  #pragma unroll
  for (int j = 0; j < 32; j += 8)
    t[ty + j][tx] = src[(size_t)(by + ty + j) * C + bx + tx];
  __syncthreads();
  #pragma unroll
  for (int j = 0; j < 32; j += 8)
    dst[(size_t)(bx + ty + j) * R + by + tx] = (__bf16)t[tx][ty + j];
}

// ---------------------------------------------------------------------------
// router: logits = x @ Wr^T, top-2 softmax, scatter (token, weight) per expert
// one wave32 per token; lanes stride H, shuffle-reduce 8 partial dots.
// ---------------------------------------------------------------------------
__global__ __launch_bounds__(256) void router_kernel(
    const float* __restrict__ x, const float* __restrict__ Wr,
    int* __restrict__ counts, int* __restrict__ idxL, float* __restrict__ wgtL) {
  int wave = threadIdx.x >> 5, lane = threadIdx.x & 31;
  int token = blockIdx.x * 8 + wave;
  float acc[EE];
  #pragma unroll
  for (int e = 0; e < EE; ++e) acc[e] = 0.0f;
  const float* xr = x + (size_t)token * HH;
  for (int h = lane; h < HH; h += 32) {
    float xv = xr[h];
    #pragma unroll
    for (int e = 0; e < EE; ++e) acc[e] += xv * Wr[e * HH + h];
  }
  #pragma unroll
  for (int e = 0; e < EE; ++e) {
    #pragma unroll
    for (int off = 16; off > 0; off >>= 1)
      acc[e] += __shfl_xor(acc[e], off, 32);
  }
  if (lane == 0) {
    int b0 = 0; float v0 = acc[0];
    #pragma unroll
    for (int e = 1; e < EE; ++e)
      if (acc[e] > v0) { v0 = acc[e]; b0 = e; }
    int b1 = -1; float v1 = -3.4e38f;
    #pragma unroll
    for (int e = 0; e < EE; ++e)
      if (e != b0 && acc[e] > v1) { v1 = acc[e]; b1 = e; }
    float e2 = __expf(v1 - v0);        // softmax over the two kept logits
    float s  = 1.0f + e2;
    float w0 = 1.0f / s, w1 = e2 / s;
    int p0 = atomicAdd(&counts[b0], 1);
    idxL[b0 * TT + p0] = token; wgtL[b0 * TT + p0] = w0;
    int p1 = atomicAdd(&counts[b1], 1);
    idxL[b1 * TT + p1] = token; wgtL[b1 * TT + p1] = w1;
  }
}

// ---------------------------------------------------------------------------
// GEMM1 (fused gate+up): act[r,:] = silu(x[tok(r)]@Wg) * (x[tok(r)]@Wu)
// block tile 128(M) x 64(N), K-step 32; 8 waves, each a 32x32 sub-tile
// (2x2 of 16x16x32 bf16 WMMA) for both gate and up accumulators.
// Double-buffered LDS fed by async global->LDS loads.
// ---------------------------------------------------------------------------
__global__ __launch_bounds__(256) void moe_gemm1(
    const __bf16* __restrict__ xb, const __bf16* __restrict__ wg,
    const __bf16* __restrict__ wu, __bf16* __restrict__ act,
    const int* __restrict__ idxL, const int* __restrict__ cntp) {
  const int cnt = *cntp;
  const int tileM = blockIdx.y, tileN = blockIdx.x;
  if (tileM * 128 >= cnt) return;

  __shared__ __bf16 Asm[2][128 * LDP];
  __shared__ __bf16 Bgs[2][64 * LDP];
  __shared__ __bf16 Bus[2][64 * LDP];

  const int tid = threadIdx.x, lane = tid & 31, wave = tid >> 5;
  const int waveM = wave & 3, waveN = wave >> 2;
  const int khalf = lane >> 4, lmod = lane & 15;

  v8f accg[2][2], accu[2][2];
  #pragma unroll
  for (int i = 0; i < 2; ++i)
    #pragma unroll
    for (int j = 0; j < 2; ++j) {
      accg[i][j] = (v8f){0,0,0,0,0,0,0,0};
      accu[i][j] = (v8f){0,0,0,0,0,0,0,0};
    }

  // global->LDS load coordinates (gathered A rows via index list)
  const int ar = tid >> 1, ac = (tid & 1) * 16;
  const int rp = tileM * 128 + ar;
  const int tok = (rp < cnt) ? idxL[rp] : idxL[tileM * 128];
  const __bf16* aSrc = xb + (size_t)tok * HH + ac;
  const int bn = tid >> 2, bc = (tid & 3) * 8;
  const __bf16* gSrc = wg + (size_t)(tileN * 64 + bn) * HH + bc;
  const __bf16* uSrc = wu + (size_t)(tileN * 64 + bn) * HH + bc;

  __bf16* aDst = &Asm[0][ar * LDP + ac];
  __bf16* gDst = &Bgs[0][bn * LDP + bc];
  __bf16* uDst = &Bus[0][bn * LDP + bc];
  const int bufStrA = 128 * LDP, bufStrB = 64 * LDP;

  // prologue: fill buffer 0
  asyncLoad16(aSrc, aDst);
  asyncLoad16(aSrc + 8, aDst + 8);
  asyncLoad16(gSrc, gDst);
  asyncLoad16(uSrc, uDst);

  int cur = 0;
  for (int k0 = 0; k0 < HH; k0 += 32, cur ^= 1) {
    waitAsync0();        // our buffer-cur chunks have landed in LDS
    __syncthreads();     // everyone's landed; prev reads of buf cur^1 done
    if (k0 + 32 < HH) {  // issue next K-step into the other buffer
      int nb = cur ^ 1, k1 = k0 + 32;
      asyncLoad16(aSrc + k1,     aDst + nb * bufStrA);
      asyncLoad16(aSrc + k1 + 8, aDst + nb * bufStrA + 8);
      asyncLoad16(gSrc + k1,     gDst + nb * bufStrB);
      asyncLoad16(uSrc + k1,     uDst + nb * bufStrB);
    }
    v16bf af[2], bgf[2], buf[2];
    #pragma unroll
    for (int i = 0; i < 2; ++i)
      af[i] = ldsFrag(&Asm[cur][(32 * waveM + 16 * i + lmod) * LDP], khalf);
    #pragma unroll
    for (int j = 0; j < 2; ++j) {
      bgf[j] = ldsFrag(&Bgs[cur][(32 * waveN + 16 * j + lmod) * LDP], khalf);
      buf[j] = ldsFrag(&Bus[cur][(32 * waveN + 16 * j + lmod) * LDP], khalf);
    }
    #pragma unroll
    for (int i = 0; i < 2; ++i)
      #pragma unroll
      for (int j = 0; j < 2; ++j) {
        accg[i][j] = __builtin_amdgcn_wmma_f32_16x16x32_bf16(
            false, af[i], false, bgf[j], (short)0, accg[i][j], false, false);
        accu[i][j] = __builtin_amdgcn_wmma_f32_16x16x32_bf16(
            false, af[i], false, buf[j], (short)0, accu[i][j], false, false);
      }
  }

  // epilogue: SiLU(gate)*up -> bf16 activation buffer (gathered row index)
  #pragma unroll
  for (int i = 0; i < 2; ++i)
    #pragma unroll
    for (int j = 0; j < 2; ++j) {
      int col = tileN * 64 + 32 * waveN + 16 * j + lmod;
      #pragma unroll
      for (int g = 0; g < 8; ++g) {
        int row = tileM * 128 + 32 * waveM + 16 * i + g + 8 * khalf;
        if (row < cnt) {
          float gv = accg[i][j][g], uv = accu[i][j][g];
          // fast SiLU: v_rcp_f32 instead of IEEE division sequence
          float h = gv * __builtin_amdgcn_rcpf(1.0f + __expf(-gv)) * uv;
          act[(size_t)row * II + col] = (__bf16)h;
        }
      }
    }
}

// ---------------------------------------------------------------------------
// GEMM2: out[tok(r),:] += w(r) * (act[r,:] @ Wd)   (K = II, N = HH)
// ---------------------------------------------------------------------------
__global__ __launch_bounds__(256) void moe_gemm2(
    const __bf16* __restrict__ act, const __bf16* __restrict__ wd,
    float* __restrict__ out, const int* __restrict__ idxL,
    const float* __restrict__ wgtL, const int* __restrict__ cntp) {
  const int cnt = *cntp;
  const int tileM = blockIdx.y, tileN = blockIdx.x;
  if (tileM * 128 >= cnt) return;

  __shared__ __bf16 Asm[2][128 * LDP];
  __shared__ __bf16 Bsm[2][64 * LDP];

  const int tid = threadIdx.x, lane = tid & 31, wave = tid >> 5;
  const int waveM = wave & 3, waveN = wave >> 2;
  const int khalf = lane >> 4, lmod = lane & 15;

  v8f acc[2][2];
  #pragma unroll
  for (int i = 0; i < 2; ++i)
    #pragma unroll
    for (int j = 0; j < 2; ++j) acc[i][j] = (v8f){0,0,0,0,0,0,0,0};

  const int ar = tid >> 1, ac = (tid & 1) * 16;
  const int rpRaw = tileM * 128 + ar;
  const int rp = (rpRaw < cnt) ? rpRaw : 0;
  const __bf16* aSrc = act + (size_t)rp * II + ac;
  const int bn = tid >> 2, bc = (tid & 3) * 8;
  const __bf16* bSrc = wd + (size_t)(tileN * 64 + bn) * II + bc;

  __bf16* aDst = &Asm[0][ar * LDP + ac];
  __bf16* bDst = &Bsm[0][bn * LDP + bc];
  const int bufStrA = 128 * LDP, bufStrB = 64 * LDP;

  asyncLoad16(aSrc, aDst);
  asyncLoad16(aSrc + 8, aDst + 8);
  asyncLoad16(bSrc, bDst);

  int cur = 0;
  for (int k0 = 0; k0 < II; k0 += 32, cur ^= 1) {
    waitAsync0();
    __syncthreads();
    if (k0 + 32 < II) {
      int nb = cur ^ 1, k1 = k0 + 32;
      asyncLoad16(aSrc + k1,     aDst + nb * bufStrA);
      asyncLoad16(aSrc + k1 + 8, aDst + nb * bufStrA + 8);
      asyncLoad16(bSrc + k1,     bDst + nb * bufStrB);
    }
    v16bf af[2], bf[2];
    #pragma unroll
    for (int i = 0; i < 2; ++i)
      af[i] = ldsFrag(&Asm[cur][(32 * waveM + 16 * i + lmod) * LDP], khalf);
    #pragma unroll
    for (int j = 0; j < 2; ++j)
      bf[j] = ldsFrag(&Bsm[cur][(32 * waveN + 16 * j + lmod) * LDP], khalf);
    #pragma unroll
    for (int i = 0; i < 2; ++i)
      #pragma unroll
      for (int j = 0; j < 2; ++j)
        acc[i][j] = __builtin_amdgcn_wmma_f32_16x16x32_bf16(
            false, af[i], false, bf[j], (short)0, acc[i][j], false, false);
  }

  // scatter-accumulate with router weight: no-return global_atomic_add_f32
  // (fire-and-forget, STOREcnt), race-free per unit and deterministic.
  #pragma unroll
  for (int i = 0; i < 2; ++i)
    #pragma unroll
    for (int j = 0; j < 2; ++j) {
      int col = tileN * 64 + 32 * waveN + 16 * j + lmod;
      #pragma unroll
      for (int g = 0; g < 8; ++g) {
        int row = tileM * 128 + 32 * waveM + 16 * i + g + 8 * khalf;
        if (row < cnt) {
          int tok = idxL[row];
          float w = wgtL[row];
          float* o = out + (size_t)tok * HH + col;
          (void)__hip_atomic_fetch_add(o, w * acc[i][j][g],
                                       __ATOMIC_RELAXED,
                                       __HIP_MEMORY_SCOPE_AGENT);
        }
      }
    }
}

// ---------------------------------------------------------------------------
// host-side orchestration
// ---------------------------------------------------------------------------
extern "C" void kernel_launch(void* const* d_in, const int* in_sizes, int n_in,
                              void* d_out, int out_size, void* d_ws, size_t ws_size,
                              hipStream_t stream) {
  (void)in_sizes; (void)n_in; (void)out_size; (void)ws_size;
  const float* x   = (const float*)d_in[0];
  const float* Wr  = (const float*)d_in[1];
  const float* Wg  = (const float*)d_in[2];
  const float* Wu  = (const float*)d_in[3];
  const float* Wd  = (const float*)d_in[4];
  const float* Wgs = (const float*)d_in[5];
  const float* Wus = (const float*)d_in[6];
  const float* Wds = (const float*)d_in[7];
  float* out = (float*)d_out;

  char* ws = (char*)d_ws;
  __bf16* xb   = (__bf16*)(ws);                       // TT*HH bf16   = 32 MB
  __bf16* act  = (__bf16*)(ws + 33554432);            // TT*II bf16   = 16 MB
  __bf16* wbg  = (__bf16*)(ws + 50331648);            // [II,HH] bf16 =  4 MB
  __bf16* wbu  = (__bf16*)(ws + 54525952);            // [II,HH] bf16 =  4 MB
  __bf16* wbd  = (__bf16*)(ws + 58720256);            // [HH,II] bf16 =  4 MB
  int*    idxL = (int*)  (ws + 62914560);             // [9][TT] int
  float*  wgtL = (float*)(ws + 63209472);             // [9][TT] f32
  int*    cnts = (int*)  (ws + 63504384);             // [9]

  zero_f32<<<(TT * HH) / 1024, 1024, 0, stream>>>(out, TT * HH);
  init_counts<<<1, 32, 0, stream>>>(cnts);
  convert_x<<<(TT * HH) / 256, 256, 0, stream>>>(x, xb);
  router_kernel<<<TT / 8, 256, 0, stream>>>(x, Wr, cnts, idxL, wgtL);
  shared_list<<<TT / 256, 256, 0, stream>>>(idxL, wgtL);

  for (int u = 0; u < 9; ++u) {
    const float *wgp, *wup, *wdp;
    if (u < 8) {
      wgp = Wg + (size_t)u * HH * II;
      wup = Wu + (size_t)u * HH * II;
      wdp = Wd + (size_t)u * II * HH;
    } else {
      wgp = Wgs; wup = Wus; wdp = Wds;
    }
    // weights -> N-major bf16 (reused buffers; stream serializes units)
    tconv<<<dim3(II / 32, HH / 32), dim3(32, 8), 0, stream>>>(wgp, wbg, HH, II);
    tconv<<<dim3(II / 32, HH / 32), dim3(32, 8), 0, stream>>>(wup, wbu, HH, II);
    tconv<<<dim3(HH / 32, II / 32), dim3(32, 8), 0, stream>>>(wdp, wbd, II, HH);
    // fused gate/up + SiLU, then weighted down-projection scatter
    moe_gemm1<<<dim3(II / 64, TT / 128), 256, 0, stream>>>(
        xb, wbg, wbu, act, idxL + u * TT, cnts + u);
    moe_gemm2<<<dim3(HH / 64, TT / 128), 256, 0, stream>>>(
        act, wbd, out, idxL + u * TT, wgtL + u * TT, cnts + u);
  }
}
